// PraxisGRU_27169963114934
// MI455X (gfx1250) — matrix operand order repaired and checked by reading, not compiled, not run on hardware
//
#include <hip/hip_runtime.h>
#include <hip/hip_bf16.h>

// ---------------------------------------------------------------------------
// CDNA5 (gfx1250) implementation: LayerNorm -> minGRU (WMMA GEMM + log-space
// scan) -> stretch -> FFN (two WMMA GEMMs, gelu) -> residual.
// Matrix math: v_wmma_f32_16x16x32_bf16 (wave32, fp32 accumulate).
// Tile staging: global_load_async_to_lds_b128 (ASYNCcnt) + double buffering.
// GEMM shapes are compile-time (template) so epilogue addressing folds to
// immediate offsets.
// ---------------------------------------------------------------------------

typedef __attribute__((ext_vector_type(16))) __bf16 v16bf;
typedef __attribute__((ext_vector_type(8)))  float  v8f;

#define BQ 4
#define SQ 4096
#define DQ 1024
#define DFFQ 4096
#define MQ (BQ * SQ)          // 16384 rows

// ---------------- helpers -------------------------------------------------

__device__ __forceinline__ float softplus_f(float x) {
    return log1pf(__expf(-fabsf(x))) + fmaxf(x, 0.0f);  // stable log(1+e^x)
}

__device__ __forceinline__ unsigned f2o(float f) {
    unsigned u = __float_as_uint(f);
    return (u & 0x80000000u) ? ~u : (u | 0x80000000u);
}
__device__ __forceinline__ float o2f(unsigned o) {
    return (o & 0x80000000u) ? __uint_as_float(o ^ 0x80000000u)
                             : __uint_as_float(~o);
}

// Low 32 bits of a generic pointer to LDS == raw LDS byte offset
__device__ __forceinline__ unsigned lds_off(const void* p) {
    return (unsigned)(unsigned long long)(uintptr_t)p;
}

// ---------------- tiny utility kernels ------------------------------------

__global__ void init_minmax_kernel(unsigned* mm) {
    mm[0] = 0xFFFFFFFFu;  // ordered +inf (min slot)
    mm[1] = 0x00000000u;  // ordered -inf (max slot)
}

__global__ void cvt_f32_bf16_kernel(const float* __restrict__ src,
                                    __bf16* __restrict__ dst, size_t n) {
    size_t i = (size_t)blockIdx.x * blockDim.x + threadIdx.x;
    if (i < n) dst[i] = (__bf16)src[i];
}

// ---------------- LayerNorm (row of D=1024) -> bf16 -----------------------

__global__ __launch_bounds__(256) void layernorm_bf16_kernel(
    const float* __restrict__ x, const float* __restrict__ w,
    const float* __restrict__ b, __bf16* __restrict__ xn) {
    const int row = blockIdx.x;
    const int t   = threadIdx.x;
    const float4 v = ((const float4*)(x + (size_t)row * DQ))[t];
    float s = v.x + v.y + v.z + v.w;
    float q = v.x * v.x + v.y * v.y + v.z * v.z + v.w * v.w;

    __shared__ float sA[256], sB[256];
    sA[t] = s; sB[t] = q;
    __syncthreads();
    for (int o = 128; o > 0; o >>= 1) {
        if (t < o) { sA[t] += sA[t + o]; sB[t] += sB[t + o]; }
        __syncthreads();
    }
    const float mu  = sA[0] * (1.0f / (float)DQ);
    const float var = sB[0] * (1.0f / (float)DQ) - mu * mu;
    const float rv  = rsqrtf(var + 1e-5f);

    const float4 wv = ((const float4*)w)[t];
    const float4 bv = ((const float4*)b)[t];
    __bf16* o4 = xn + (size_t)row * DQ + t * 4;
    o4[0] = (__bf16)((v.x - mu) * rv * wv.x + bv.x);
    o4[1] = (__bf16)((v.y - mu) * rv * wv.y + bv.y);
    o4[2] = (__bf16)((v.z - mu) * rv * wv.z + bv.z);
    o4[3] = (__bf16)((v.w - mu) * rv * wv.w + bv.w);
}

// ---------------- WMMA GEMM: C[M,N] = A[M,K] * B[N,K]^T -------------------
// Double-buffered LDS, tiles staged with global_load_async_to_lds_b128.
// N, K compile-time so all addressing folds to immediates.
// MODE 0: fp32 store (hg) | MODE 1: +bias, gelu, bf16 | MODE 2: +bias+resid

template <int MODE, int N, int K>
__global__ __launch_bounds__(256, 2) void gemm_bf16_wmma(
    const __bf16* __restrict__ A, const __bf16* __restrict__ Bw,
    const float* __restrict__ bias, const float* __restrict__ resid,
    float* __restrict__ outF, __bf16* __restrict__ outH) {
    constexpr int BM = 128, BN = 128, BK = 32, LDP = 40;  // +8 bf16 pad
    __shared__ alignas(16) __bf16 Als[2][BM][LDP];
    __shared__ alignas(16) __bf16 Bls[2][BN][LDP];

    const int t    = threadIdx.x;
    const int wave = t >> 5;
    const int lane = t & 31;
    const int l16  = lane & 15;
    const int lh   = lane >> 4;
    const int wm   = wave & 3;   // 4 waves in M -> 32 rows each
    const int wn   = wave >> 2;  // 2 waves in N -> 64 cols each
    const int m0   = blockIdx.y * BM;
    const int n0   = blockIdx.x * BN;
    const int lrow = t >> 1;
    const int lcol = (t & 1) * 16;  // each thread stages 16 bf16 = 32B

    const v8f vzero = {0.f, 0.f, 0.f, 0.f, 0.f, 0.f, 0.f, 0.f};
    v8f acc[2][4];
#pragma unroll
    for (int i = 0; i < 2; ++i)
#pragma unroll
        for (int j = 0; j < 4; ++j) acc[i][j] = vzero;

    union Frag { v16bf v; unsigned u[8]; };

    // per-thread global base addresses (K-tile advances by constant stride)
    const __bf16* gA = A + (size_t)(m0 + lrow) * K + lcol;
    const __bf16* gB = Bw + (size_t)(n0 + lrow) * K + lcol;

    // async memory->LDS staging of one 128x32 A tile + 128x32 B tile
    auto issue_tile = [&](int kt, int buf) {
        const unsigned long long ga =
            (unsigned long long)(uintptr_t)(gA + kt * BK);
        const unsigned long long gb =
            (unsigned long long)(uintptr_t)(gB + kt * BK);
        const unsigned la = lds_off(&Als[buf][lrow][lcol]);
        const unsigned lb = lds_off(&Bls[buf][lrow][lcol]);
        asm volatile(
            "global_load_async_to_lds_b128 %0, %2, off\n\t"
            "global_load_async_to_lds_b128 %0, %2, off offset:16\n\t"
            "global_load_async_to_lds_b128 %1, %3, off\n\t"
            "global_load_async_to_lds_b128 %1, %3, off offset:16"
            :: "v"(la), "v"(lb), "v"(ga), "v"(gb)
            : "memory");
    };

    constexpr int nk = K / BK;
    issue_tile(0, 0);
    asm volatile("s_wait_asynccnt 0x0" ::: "memory");
    __syncthreads();

    for (int kt = 0; kt < nk; ++kt) {
        const int buf = kt & 1;
        if (kt + 1 < nk) issue_tile(kt + 1, buf ^ 1);  // prefetch next tile

        // A fragments: 16x32 bf16 per wave-frag (ISA 7.12.2 layout)
        Frag af[2];
#pragma unroll
        for (int mf = 0; mf < 2; ++mf) {
            const int mr = wm * 32 + mf * 16 + l16;
#pragma unroll
            for (int v = 0; v < 8; ++v) {
                const int kk = (v < 4) ? (8 * lh + 2 * v)
                                       : (16 + 8 * lh + 2 * (v - 4));
                af[mf].u[v] = *(const unsigned*)&Als[buf][mr][kk];
            }
        }
        // B fragments: 32x16 bf16, rows striped across lanes
        Frag bfr[4];
#pragma unroll
        for (int nf = 0; nf < 4; ++nf) {
            const int nr = wn * 64 + nf * 16 + l16;
#pragma unroll
            for (int v = 0; v < 8; ++v) {
                const int kk = (v < 4) ? (8 * lh + 2 * v)
                                       : (16 + 8 * lh + 2 * (v - 4));
                bfr[nf].u[v] = *(const unsigned*)&Bls[buf][nr][kk];
            }
        }

#pragma unroll
        for (int mf = 0; mf < 2; ++mf)
#pragma unroll
            for (int nf = 0; nf < 4; ++nf)
                acc[mf][nf] = __builtin_amdgcn_wmma_f32_16x16x32_bf16(
                    false, af[mf].v, false, bfr[nf].v, (short)0, acc[mf][nf],
                    false, false);

        if (kt + 1 < nk)
            asm volatile("s_wait_asynccnt 0x0" ::: "memory");
        __syncthreads();
    }

    // epilogue: C layout -> row = 8*lh + r, col = l16 within 16x16 tile.
    // Base row/col per (mf,nf) is constant-strided; r advances rows by N
    // (compile-time) so stores fold to base + immediate offsets.
#pragma unroll
    for (int mf = 0; mf < 2; ++mf) {
#pragma unroll
        for (int nf = 0; nf < 4; ++nf) {
            const int mB = m0 + wm * 32 + mf * 16 + lh * 8;
            const int n  = n0 + wn * 64 + nf * 16 + l16;
            const size_t base = (size_t)mB * N + n;
#pragma unroll
            for (int r = 0; r < 8; ++r) {
                float v = acc[mf][nf][r];
                if (MODE == 1) {
                    v += bias[n];
                    const float u = v;
                    v = 0.5f * u *
                        (1.0f + tanhf(0.7978845608f * (u + 0.044715f * u * u * u)));
                    outH[base + (size_t)r * N] = (__bf16)v;
                } else if (MODE == 2) {
                    v += bias[n] + resid[base + (size_t)r * N];
                    outF[base + (size_t)r * N] = v;
                } else {
                    outF[base + (size_t)r * N] = v;
                }
            }
        }
    }
}

// ---------------- gate math (in place on hg[M, 2D]) -----------------------
// hidden slot <- log_z + log_g(hidden) ; gate slot <- log(1-z)

__global__ void gru_gates_kernel(float* __restrict__ hg) {
    const size_t i = (size_t)blockIdx.x * blockDim.x + threadIdx.x;
    if (i >= (size_t)MQ * DQ) return;
    const size_t m = i / DQ;
    const int    d = (int)(i % DQ);
    float* p = hg + m * (size_t)(2 * DQ);
    const float hid  = p[d];
    const float gate = p[d + DQ];
    const float lc = -softplus_f(gate);        // log(1-z)
    const float lz = -softplus_f(-gate);       // log(z)
    const float lg = (hid >= 0.0f) ? __logf(hid + 0.5f) : -softplus_f(-hid);
    p[d]      = lz + lg;                       // log_values
    p[d + DQ] = lc;                            // log_coeffs
}

// ---------------- streaming log-space scan + global min/max ---------------

__global__ __launch_bounds__(256) void gru_scan_minmax_kernel(
    const float* __restrict__ hg, float* __restrict__ out,
    unsigned* __restrict__ mm) {
    const int ch = blockIdx.x * blockDim.x + threadIdx.x;  // 0 .. B*D-1
    const int b  = ch / DQ;
    const int d  = ch % DQ;
    size_t base = ((size_t)b * SQ) * (2 * DQ) + d;
    size_t ob   = ((size_t)b * SQ) * DQ + d;
    float a = 0.0f, mrun = -__builtin_inff(), srun = 0.0f;
    float mn = __builtin_inff(), mx = -__builtin_inff();
    for (int s = 0; s < SQ; ++s) {
        const float lv = hg[base];        // log_values
        const float lc = hg[base + DQ];   // log_coeffs
        a += lc;                          // a_star (cumsum)
        const float xv = lv - a;          // cumlogsumexp argument
        if (xv > mrun) {
            srun = srun * __expf(mrun - xv) + 1.0f;
            mrun = xv;
        } else {
            srun += __expf(xv - mrun);
        }
        const float h = __expf(a + mrun + __logf(srun));
        out[ob] = h;
        mn = fminf(mn, h);
        mx = fmaxf(mx, h);
        base += (size_t)(2 * DQ);
        ob   += DQ;
    }
    __shared__ unsigned smn[256], smx[256];
    const int t = threadIdx.x;
    smn[t] = f2o(mn);
    smx[t] = f2o(mx);
    __syncthreads();
    for (int o = 128; o > 0; o >>= 1) {
        if (t < o) {
            smn[t] = (smn[t + o] < smn[t]) ? smn[t + o] : smn[t];
            smx[t] = (smx[t + o] > smx[t]) ? smx[t + o] : smx[t];
        }
        __syncthreads();
    }
    if (t == 0) {
        atomicMin(&mm[0], smn[0]);
        atomicMax(&mm[1], smx[0]);
    }
}

// ---------------- stretch_min -> bf16 -------------------------------------

__global__ void stretch_bf16_kernel(const float* __restrict__ in,
                                    const unsigned* __restrict__ mm,
                                    __bf16* __restrict__ outH, size_t n) {
    const size_t i = (size_t)blockIdx.x * blockDim.x + threadIdx.x;
    if (i >= n) return;
    const float mn = o2f(mm[0]);
    const float mx = o2f(mm[1]);
    const float p  = (in[i] - mn) / (mx - mn);
    outH[i] = (__bf16)(-1.0f + p * (mx + 1.0f));
}

// ---------------------------------------------------------------------------

extern "C" void kernel_launch(void* const* d_in, const int* in_sizes, int n_in,
                              void* d_out, int out_size, void* d_ws, size_t ws_size,
                              hipStream_t stream) {
    (void)in_sizes; (void)n_in; (void)out_size; (void)ws_size;
    const float* x    = (const float*)d_in[0];
    const float* ln_w = (const float*)d_in[1];
    const float* ln_b = (const float*)d_in[2];
    const float* w_hg = (const float*)d_in[3];
    const float* w1   = (const float*)d_in[4];
    const float* b1   = (const float*)d_in[5];
    const float* w2   = (const float*)d_in[6];
    const float* b2   = (const float*)d_in[7];
    float* outp = (float*)d_out;

    // workspace carve (byte offsets, all 256B-aligned)
    char* ws = (char*)d_ws;
    __bf16* whg_bf = (__bf16*)(ws + 0);                         //  4 MiB
    __bf16* w1_bf  = (__bf16*)(ws + (size_t)4194304);           //  8 MiB
    __bf16* w2_bf  = (__bf16*)(ws + (size_t)12582912);          //  8 MiB
    __bf16* xn_bf  = (__bf16*)(ws + (size_t)20971520);          // 32 MiB (reused as s_bf)
    float*  hg     = (float*) (ws + (size_t)54525952);          // 128 MiB (reused as h_bf)
    float*  gru    = (float*) (ws + (size_t)188743680);         // 64 MiB
    unsigned* mm   = (unsigned*)(ws + (size_t)255852544);
    __bf16* s_bf   = xn_bf;            // alias: xn dead after GEMM1
    __bf16* h_bf   = (__bf16*)hg;      // alias: hg dead after scan (equal size)

    init_minmax_kernel<<<1, 1, 0, stream>>>(mm);

    // weights -> bf16
    cvt_f32_bf16_kernel<<<(2 * DQ * DQ) / 256, 256, 0, stream>>>(w_hg, whg_bf,
                                                                 (size_t)2 * DQ * DQ);
    cvt_f32_bf16_kernel<<<(DFFQ * DQ) / 256, 256, 0, stream>>>(w1, w1_bf,
                                                               (size_t)DFFQ * DQ);
    cvt_f32_bf16_kernel<<<(DQ * DFFQ) / 256, 256, 0, stream>>>(w2, w2_bf,
                                                               (size_t)DQ * DFFQ);

    // 1. LayerNorm
    layernorm_bf16_kernel<<<MQ, 256, 0, stream>>>(x, ln_w, ln_b, xn_bf);

    // 2. hg = xn @ w_hg^T  (16384 x 2048 x 1024)
    gemm_bf16_wmma<0, 2 * DQ, DQ><<<dim3(2 * DQ / 128, MQ / 128), 256, 0, stream>>>(
        xn_bf, whg_bf, nullptr, nullptr, hg, nullptr);

    // 3. gate math (in place)
    gru_gates_kernel<<<((size_t)MQ * DQ) / 256, 256, 0, stream>>>(hg);

    // 4. log-space scan along S + global min/max
    gru_scan_minmax_kernel<<<(BQ * DQ) / 256, 256, 0, stream>>>(hg, gru, mm);

    // 5. stretch -> bf16 s
    stretch_bf16_kernel<<<((size_t)MQ * DQ) / 256, 256, 0, stream>>>(
        gru, mm, s_bf, (size_t)MQ * DQ);

    // 6. h = gelu(s @ w1^T + b1)  (16384 x 4096 x 1024)
    gemm_bf16_wmma<1, DFFQ, DQ><<<dim3(DFFQ / 128, MQ / 128), 256, 0, stream>>>(
        s_bf, w1_bf, b1, nullptr, nullptr, h_bf);

    // 7. out = h @ w2^T + b2 + x  (16384 x 1024 x 4096)
    gemm_bf16_wmma<2, DQ, DFFQ><<<dim3(DQ / 128, MQ / 128), 256, 0, stream>>>(
        h_bf, w2_bf, b2, x, outp, nullptr);
}
